// RelativePositionSelfAttention_17660905521872
// MI455X (gfx1250) — matrix operand order repaired and checked by reading, not compiled, not run on hardware
//
#include <hip/hip_runtime.h>
#include <hip/hip_bf16.h>

// ---------------- problem constants ----------------
#define B_ 4
#define S_ 2048
#define E_ 1024
#define H_ 16
#define D_ 64
#define MREL 32
#define TOKENS (B_ * S_)   // 8192
#define QKV_LD (3 * E_)    // 3072

typedef __attribute__((ext_vector_type(16))) _Float16 v16h;
typedef __attribute__((ext_vector_type(8)))  _Float16 v8h;
typedef __attribute__((ext_vector_type(4)))  _Float16 v4h;
typedef __attribute__((ext_vector_type(8)))  float    v8f;
typedef __attribute__((ext_vector_type(4)))  float    v4fx;

// D = A(16x32 f16) * B(32x16 f16) + C(16x16 f32), wave32
__device__ inline v8f wmma16(v16h a, v16h b, v8f c) {
  return __builtin_amdgcn_wmma_f32_16x16x32_f16(
      /*neg_a=*/false, a, /*neg_b=*/false, b,
      /*c_mod=*/(short)0, c, /*reuse_a=*/false, /*reuse_b=*/false);
}

__device__ inline v16h cat8(v8h lo, v8h hi) {
  v16h r;
#pragma unroll
  for (int i = 0; i < 8; i++) { r[i] = lo[i]; r[8 + i] = hi[i]; }
  return r;
}

// A-fragment (16x32, K-contiguous row): rowp = this lane's row with k0 folded in.
// ISA layout: half=0 -> K 0..7 and 16..23 ; half=1 -> K 8..15 and 24..31
__device__ inline v16h afrag_f16(const _Float16* rowp, int half) {
  v8h lo = *(const v8h*)(rowp + half * 8);
  v8h hi = *(const v8h*)(rowp + half * 8 + 16);
  return cat8(lo, hi);
}

// B-fragment (32x16) sourced K-contiguously (computing X * Wt):
// lane n holds K = half*8 .. half*8+7 and 16+half*8 .. 16+half*8+7 of column n.
__device__ inline v16h bfrag_f16(const _Float16* rowp, int half) {
  v8h lo = *(const v8h*)(rowp + half * 8);
  v8h hi = *(const v8h*)(rowp + half * 8 + 16);
  return cat8(lo, hi);
}

// ---------------- kernel 0: mean over heads of rel_bias ----------------
__global__ void bias_mean_kernel(const float* __restrict__ rb, float* __restrict__ mb) {
  int i = threadIdx.x;
  if (i < 2 * MREL + 1) {
    float s = 0.f;
#pragma unroll
    for (int hh = 0; hh < H_; hh++) s += rb[i * H_ + hh];
    mb[i] = s * (1.0f / H_);
  }
}

// ---------------- kernel 1: bulk fp32 -> f16 conversion ----------------
__global__ __launch_bounds__(256) void cvt_f16_kernel(const float* __restrict__ src,
                                                      _Float16* __restrict__ dst, int n) {
  int i = (blockIdx.x * 256 + threadIdx.x) * 4;
  if (i < n) {
    v4fx f = *(const v4fx*)(src + i);
    v4h o;
    o[0] = (_Float16)f[0]; o[1] = (_Float16)f[1];
    o[2] = (_Float16)f[2]; o[3] = (_Float16)f[3];
    *(v4h*)(dst + i) = o;
  }
}

// ---------------- shared GEMM mainloop: 64x64 output per wave ----------------
// A: f16 [*, lda] row-major at row block m0 (pre-offset), B: f16 [*, ldb] at col block n0.
// 16 WMMAs per 32-wide k step, no LDS, no barriers; acc[i][j] = Mtile i, Ntile j.
__device__ inline void gemm_core64(const _Float16* __restrict__ A, int lda,
                                   const _Float16* __restrict__ Bm, int ldb,
                                   int K, int m, int half, v8f (&acc)[4][4]) {
  const _Float16* ar[4];
  const _Float16* br[4];
#pragma unroll
  for (int t = 0; t < 4; t++) {
    ar[t] = A + (size_t)(t * 16 + m) * lda;
    br[t] = Bm + (size_t)(t * 16 + m) * ldb;
  }
  for (int k0 = 0; k0 < K; k0 += 32) {
    v16h a[4], b[4];
#pragma unroll
    for (int t = 0; t < 4; t++) a[t] = afrag_f16(ar[t] + k0, half);
#pragma unroll
    for (int t = 0; t < 4; t++) b[t] = bfrag_f16(br[t] + k0, half);
#pragma unroll
    for (int i = 0; i < 4; i++)
#pragma unroll
      for (int j = 0; j < 4; j++)
        acc[i][j] = wmma16(a[i], b[j], acc[i][j]);
  }
}

// ---------------- kernel 2: qkv = x16 @ w1_16^T + b1 (f16 out, Q pre-scaled) ------
// block: 8 waves, each a 64x64 tile; block covers 64(M) x 512(N).
__global__ __launch_bounds__(256) void qkv_gemm_kernel(
    const _Float16* __restrict__ x16, const _Float16* __restrict__ w16,
    const float* __restrict__ bias, _Float16* __restrict__ qkv) {
  const int wave = threadIdx.x >> 5;
  const int lane = threadIdx.x & 31;
  const int m = lane & 15, half = lane >> 4;
  const int m0 = blockIdx.y * 64;
  const int n0 = blockIdx.x * 512 + wave * 64;
  v8f acc[4][4] = {};
  gemm_core64(x16 + (size_t)m0 * E_, E_, w16 + (size_t)n0 * E_, E_, E_, m, half, acc);
#pragma unroll
  for (int j = 0; j < 4; j++) {
    int col = n0 + j * 16 + m;
    float bv = bias[col];
    float sc = (col < E_) ? 0.125f : 1.0f;  // fold D^-0.5 into Q
#pragma unroll
    for (int i = 0; i < 4; i++)
#pragma unroll
      for (int r = 0; r < 8; r++) {
        int row = m0 + i * 16 + r + 8 * half;  // C layout
        qkv[(size_t)row * QKV_LD + col] = (_Float16)((acc[i][j][r] + bv) * sc);
      }
  }
}

// ---------------- kernel 3: Vt[b,h,d,s] = V[b,s,h,d] (f16 transpose via LDS) ------
__global__ __launch_bounds__(256) void v_transpose_kernel(
    const _Float16* __restrict__ qkv, _Float16* __restrict__ vt) {
  __shared__ _Float16 tile[64 * 72];  // 64 keys x 64 dims (+pad)
  const int h = blockIdx.y & (H_ - 1);
  const int bb = blockIdx.y >> 4;
  const int k0 = blockIdx.x * 64;
  const size_t tokbase = (size_t)bb * S_;
  // load 64x64 (keys x dims), coalesced along dims
  {
    int kr = threadIdx.x >> 2;           // key row 0..63
    int dc = (threadIdx.x & 3) * 16;     // dim chunk
    const _Float16* src = qkv + (tokbase + k0 + kr) * QKV_LD + 2 * E_ + h * D_ + dc;
    *(v8h*)(&tile[kr * 72 + dc]) = *(const v8h*)src;
    *(v8h*)(&tile[kr * 72 + dc + 8]) = *(const v8h*)(src + 8);
  }
  __syncthreads();
  // store transposed, coalesced along keys
  {
    int d = threadIdx.x >> 2;            // dim 0..63
    int kc = (threadIdx.x & 3) * 16;     // key chunk
    v8h o0, o1;
#pragma unroll
    for (int i = 0; i < 8; i++) {
      o0[i] = tile[(kc + i) * 72 + d];
      o1[i] = tile[(kc + 8 + i) * 72 + d];
    }
    _Float16* dst = vt + ((size_t)(bb * H_ + h) * D_ + d) * S_ + k0 + kc;
    *(v8h*)dst = o0;
    *(v8h*)(dst + 8) = o1;
  }
}

// ---------------- kernel 4: flash attention, one wave per 16-query tile ----------
__global__ __launch_bounds__(32) void attn_kernel(
    const _Float16* __restrict__ qkv, const _Float16* __restrict__ vt,
    const float* __restrict__ mb, _Float16* __restrict__ aout) {
  __shared__ _Float16 Ps[16 * 32];  // P: C-layout -> A-layout relay
  const int lane = threadIdx.x;
  const int m = lane & 15, half = lane >> 4;
  const int qt = blockIdx.x & (S_ / 16 - 1);
  const int h  = (blockIdx.x >> 7) & (H_ - 1);
  const int bb = blockIdx.x >> 11;
  const int q0 = qt * 16;
  const size_t tokbase = (size_t)bb * S_;
  const _Float16* Qb = qkv + (tokbase + q0) * QKV_LD + h * D_;
  const _Float16* Kb = qkv + tokbase * QKV_LD + E_ + h * D_;
  const _Float16* Vtb = vt + (size_t)(bb * H_ + h) * D_ * S_;

  // Q fragments for both 32-channel chunks of D=64 (pre-scaled by D^-0.5)
  v16h qa0 = afrag_f16(Qb + (size_t)m * QKV_LD + 0, half);
  v16h qa1 = afrag_f16(Qb + (size_t)m * QKV_LD + 32, half);

  float mrow[8], lrow[8];
  v8f acc[4] = {};
#pragma unroll
  for (int r = 0; r < 8; r++) { mrow[r] = -1e30f; lrow[r] = 0.f; }

  for (int j0 = 0; j0 < S_; j0 += 32) {
    if (j0 + 32 < S_) {
      __builtin_prefetch(Kb + (size_t)(j0 + 32 + lane) * QKV_LD, 0, 0);
      __builtin_prefetch(Vtb + (size_t)(lane * 2) * S_ + j0 + 32, 0, 0);
    }
    // scores: 16 queries x 32 keys, contract over D=64
    const _Float16* kr0 = Kb + (size_t)(j0 + m) * QKV_LD;       // keys j0..j0+15
    const _Float16* kr1 = Kb + (size_t)(j0 + 16 + m) * QKV_LD;  // keys j0+16..+31
    v8f s0 = {}, s1 = {};
    s0 = wmma16(qa0, bfrag_f16(kr0 + 0, half), s0);
    s0 = wmma16(qa1, bfrag_f16(kr0 + 32, half), s0);
    s1 = wmma16(qa0, bfrag_f16(kr1 + 0, half), s1);
    s1 = wmma16(qa1, bfrag_f16(kr1 + 32, half), s1);

    // rel-pos bias + online softmax (each row owned by one 16-lane half)
#pragma unroll
    for (int r = 0; r < 8; r++) {
      int qi = q0 + r + 8 * half;
      int d0 = qi - (j0 + m);
      int d1 = d0 - 16;
      d0 = min(max(d0, -MREL), MREL) + MREL;
      d1 = min(max(d1, -MREL), MREL) + MREL;
      float v0 = s0[r] + mb[d0];
      float v1 = s1[r] + mb[d1];
      float rowmax = fmaxf(v0, v1);
#pragma unroll
      for (int off = 1; off < 16; off <<= 1)
        rowmax = fmaxf(rowmax, __shfl_xor(rowmax, off, 32));
      float Mn = fmaxf(mrow[r], rowmax);
      float alpha = __expf(mrow[r] - Mn);
      mrow[r] = Mn;
      float p0 = __expf(v0 - Mn);
      float p1 = __expf(v1 - Mn);
      float rs = p0 + p1;
#pragma unroll
      for (int off = 1; off < 16; off <<= 1)
        rs += __shfl_xor(rs, off, 32);
      lrow[r] = lrow[r] * alpha + rs;
      acc[0][r] *= alpha; acc[1][r] *= alpha;
      acc[2][r] *= alpha; acc[3][r] *= alpha;
      int rr = r + 8 * half;
      Ps[rr * 32 + m] = (_Float16)p0;
      Ps[rr * 32 + 16 + m] = (_Float16)p1;
    }
    __syncthreads();

    // O += P(16x32) * V(32x64); V fragments K-contiguous from Vt
    v16h pa = afrag_f16(&Ps[m * 32], half);
#pragma unroll
    for (int t = 0; t < 4; t++) {
      v16h vb = bfrag_f16(Vtb + (size_t)(t * 16 + m) * S_ + j0, half);
      acc[t] = wmma16(pa, vb, acc[t]);
    }
    __syncthreads();
  }

  // normalize and store f16 [token, h*64 + dim]
#pragma unroll
  for (int r = 0; r < 8; r++) {
    int row = q0 + r + 8 * half;
    float inv = 1.f / lrow[r];
    size_t base = (tokbase + row) * E_ + h * D_;
    aout[base + 0 + m]  = (_Float16)(acc[0][r] * inv);
    aout[base + 16 + m] = (_Float16)(acc[1][r] * inv);
    aout[base + 32 + m] = (_Float16)(acc[2][r] * inv);
    aout[base + 48 + m] = (_Float16)(acc[3][r] * inv);
  }
}

// ---------------- kernel 5: out = attn @ w2_16^T + b2 (f32 out) ----------------
__global__ __launch_bounds__(256) void out_gemm_kernel(
    const _Float16* __restrict__ a16, const _Float16* __restrict__ w16,
    const float* __restrict__ bias, float* __restrict__ out) {
  const int wave = threadIdx.x >> 5;
  const int lane = threadIdx.x & 31;
  const int m = lane & 15, half = lane >> 4;
  const int m0 = blockIdx.y * 64;
  const int n0 = blockIdx.x * 512 + wave * 64;
  v8f acc[4][4] = {};
  gemm_core64(a16 + (size_t)m0 * E_, E_, w16 + (size_t)n0 * E_, E_, E_, m, half, acc);
#pragma unroll
  for (int j = 0; j < 4; j++) {
    int col = n0 + j * 16 + m;
    float bv = bias[col];
#pragma unroll
    for (int i = 0; i < 4; i++)
#pragma unroll
      for (int r = 0; r < 8; r++) {
        int row = m0 + i * 16 + r + 8 * half;
        out[(size_t)row * E_ + col] = acc[i][j][r] + bv;
      }
  }
}

// ---------------- launch ----------------
extern "C" void kernel_launch(void* const* d_in, const int* in_sizes, int n_in,
                              void* d_out, int out_size, void* d_ws, size_t ws_size,
                              hipStream_t stream) {
  (void)in_sizes; (void)n_in; (void)out_size; (void)ws_size;
  const float* x  = (const float*)d_in[0];
  const float* w1 = (const float*)d_in[1];
  const float* b1 = (const float*)d_in[2];
  const float* w2 = (const float*)d_in[3];
  const float* b2 = (const float*)d_in[4];
  const float* rb = (const float*)d_in[5];
  float* out = (float*)d_out;

  char* ws = (char*)d_ws;
  size_t off = 0;
  _Float16* qkv  = (_Float16*)(ws + off); off += (size_t)TOKENS * QKV_LD * 2;  // 50.3 MB
  _Float16* att  = (_Float16*)(ws + off); off += (size_t)TOKENS * E_ * 2;      // 16.8 MB
  _Float16* x16  = (_Float16*)(ws + off); off += (size_t)TOKENS * E_ * 2;      // 16.8 MB
  _Float16* w116 = (_Float16*)(ws + off); off += (size_t)3 * E_ * E_ * 2;      //  6.3 MB
  _Float16* w216 = (_Float16*)(ws + off); off += (size_t)E_ * E_ * 2;          //  2.1 MB
  _Float16* vt   = (_Float16*)(ws + off); off += (size_t)TOKENS * E_ * 2;      // 16.8 MB
  float*    mb   = (float*)(ws + off);

  bias_mean_kernel<<<1, 128, 0, stream>>>(rb, mb);
  cvt_f16_kernel<<<TOKENS * E_ / 1024, 256, 0, stream>>>(x, x16, TOKENS * E_);
  cvt_f16_kernel<<<3 * E_ * E_ / 1024, 256, 0, stream>>>(w1, w116, 3 * E_ * E_);
  cvt_f16_kernel<<<E_ * E_ / 1024, 256, 0, stream>>>(w2, w216, E_ * E_);

  qkv_gemm_kernel<<<dim3(QKV_LD / 512, TOKENS / 64), 256, 0, stream>>>(x16, w116, b1, qkv);
  v_transpose_kernel<<<dim3(S_ / 64, B_ * H_), 256, 0, stream>>>(qkv, vt);
  attn_kernel<<<dim3(B_ * H_ * (S_ / 16)), 32, 0, stream>>>(qkv, vt, mb, att);
  out_gemm_kernel<<<dim3(E_ / 512, TOKENS / 64), 256, 0, stream>>>(att, w216, b2, out);
}
